// NoisyGate_18167711662082
// MI455X (gfx1250) — matrix operand order repaired and checked by reading, not compiled, not run on hardware
//
#include <hip/hip_runtime.h>
#include <hip/hip_bf16.h>
#include <math.h>

// ---------------------------------------------------------------------------
// NoisyGate MoE router for MI455X (gfx1250, wave32, WMMA)
//   clean = inp @ Wg ; std = softplus(inp @ Wn) + 0.01 ; noisy = clean + z*std
//   top-3 -> gates(softmax top-2), importance/load scatter, cv^2 loss
// ---------------------------------------------------------------------------

typedef __attribute__((ext_vector_type(16))) __bf16 v16bf;
typedef __attribute__((ext_vector_type(8)))  float  v8f;

#define NROWS 32768
#define DDIM  1024
#define EEXP  64

// d_ws layout:
//   [0      , 128K) : bf16 w_gate  transposed [E][D]
//   [128K   , 256K) : bf16 w_noise transposed [E][D]
//   [256K   , +256) : float importance[64]
//   [256K+256,+512) : float load[64]

// ---------------------------------------------------------------------------
// Kernel 0: transpose + convert weights to bf16, zero accumulators.
// ---------------------------------------------------------------------------
__global__ void prep_kernel(const float* __restrict__ w_gate,
                            const float* __restrict__ w_noise,
                            __bf16* __restrict__ wsG,
                            __bf16* __restrict__ wsN,
                            float* __restrict__ acc /* 128 floats: imp+load */) {
    int idx = blockIdx.x * blockDim.x + threadIdx.x;
    if (idx < 2 * EEXP) acc[idx] = 0.0f;
    if (idx < DDIM * EEXP) {
        int k = idx / EEXP;        // K index (row of W)
        int n = idx % EEXP;        // expert index (col of W)
        wsG[(size_t)n * DDIM + k] = (__bf16)w_gate[idx];
        wsN[(size_t)n * DDIM + k] = (__bf16)w_noise[idx];
    }
}

// ---------------------------------------------------------------------------
// Kernel 1: fused dual-GEMM (WMMA bf16) + gating epilogue.
// One wave -> 16 rows; 8 waves/block -> 128 rows/block; 256 blocks.
// ---------------------------------------------------------------------------
__global__ __launch_bounds__(256) void moe_gate_kernel(
    const float* __restrict__ inp,     // [N, D]
    const float* __restrict__ noise,   // [N, E]
    const __bf16* __restrict__ wsG,    // [E, D] bf16
    const __bf16* __restrict__ wsN,    // [E, D] bf16
    float* __restrict__ outIdx,        // [N*2] (indices as float)
    float* __restrict__ outGate,       // [N*2]
    float* __restrict__ gImp,          // [E]
    float* __restrict__ gLoad)         // [E]
{
    __shared__ float ldsClean[8][16][EEXP + 1];  // +1 pad: stride 65 dwords
    __shared__ float ldsStd  [8][16][EEXP + 1];
    __shared__ float impPart[EEXP];
    __shared__ float loadPart[EEXP];

    const int tid  = threadIdx.x;
    const int w    = tid >> 5;       // wave in block
    const int lane = tid & 31;
    const int m    = lane & 15;      // row (A) / column (B,C) within tile
    const int hi   = lane >> 4;      // lane half-group
    const int row0 = blockIdx.x * 128 + w * 16;

    if (tid < EEXP) { impPart[tid] = 0.0f; loadPart[tid] = 0.0f; }

    v8f accG[4], accN[4];
#pragma unroll
    for (int t = 0; t < 4; ++t) {
        accG[t] = (v8f){0.f,0.f,0.f,0.f,0.f,0.f,0.f,0.f};
        accN[t] = (v8f){0.f,0.f,0.f,0.f,0.f,0.f,0.f,0.f};
    }

    // ---- K loop: D=1024 in chunks of 32 (bf16 WMMA K) -------------------
    const float* __restrict__ arow = inp + (size_t)(row0 + m) * DDIM;
    for (int k0 = 0; k0 < DDIM; k0 += 32) {
        // A fragment, contiguous K order (matches B):
        //   elems 0..15 : K = hi*16 + e
        const float4* pa = (const float4*)(arow + k0 + hi * 16);
        float4 f0 = pa[0];
        float4 f1 = pa[1];
        const float4* pb = (const float4*)(arow + k0 + hi * 16 + 8);
        float4 f2 = pb[0];
        float4 f3 = pb[1];
        v16bf a;
        a[0]=(__bf16)f0.x; a[1]=(__bf16)f0.y; a[2]=(__bf16)f0.z; a[3]=(__bf16)f0.w;
        a[4]=(__bf16)f1.x; a[5]=(__bf16)f1.y; a[6]=(__bf16)f1.z; a[7]=(__bf16)f1.w;
        a[8]=(__bf16)f2.x; a[9]=(__bf16)f2.y; a[10]=(__bf16)f2.z; a[11]=(__bf16)f2.w;
        a[12]=(__bf16)f3.x; a[13]=(__bf16)f3.y; a[14]=(__bf16)f3.z; a[15]=(__bf16)f3.w;

#pragma unroll
        for (int t = 0; t < 4; ++t) {
            // B fragment (32x16): lane col n=m, elems e: K = hi*16 + e.
            // ws is [E][D] so this is 16 contiguous bf16 = 32 bytes.
            const v16bf* bg = (const v16bf*)(wsG + (size_t)(t * 16 + m) * DDIM + k0 + hi * 16);
            v16bf b0 = *bg;
            accG[t] = __builtin_amdgcn_wmma_f32_16x16x32_bf16(
                false, a, false, b0, (short)0, accG[t], false, false);
            const v16bf* bn = (const v16bf*)(wsN + (size_t)(t * 16 + m) * DDIM + k0 + hi * 16);
            v16bf b1 = *bn;
            accN[t] = __builtin_amdgcn_wmma_f32_16x16x32_bf16(
                false, a, false, b1, (short)0, accN[t], false, false);
        }
    }

    // ---- spill logits to LDS (C layout: VGPR r, lane -> row hi*8+r, col t*16+m)
#pragma unroll
    for (int t = 0; t < 4; ++t) {
#pragma unroll
        for (int r = 0; r < 8; ++r) {
            const int rr = hi * 8 + r;
            const int e  = t * 16 + m;
            ldsClean[w][rr][e] = accG[t][r];
            float x  = accN[t][r];
            float sp = (x > 20.0f) ? x : log1pf(expf(x));   // softplus
            ldsStd[w][rr][e] = sp + 0.01f;                  // + NOISE_EPS
        }
    }
    __syncthreads();

    // ---- per-row gating: lanes 0..15 each own one of the wave's 16 rows
    if (lane < 16) {
        const int   r    = lane;
        const int   grow = row0 + r;
        const float* __restrict__ nrow = noise + (size_t)grow * EEXP;

        float t0 = -INFINITY, t1 = -INFINITY, t2 = -INFINITY;
        int   i0 = 0, i1 = 0;
        for (int e = 0; e < EEXP; ++e) {
            float nl = ldsClean[w][r][e] + nrow[e] * ldsStd[w][r][e];
            if (nl > t0)      { t2 = t1; t1 = t0; i1 = i0; t0 = nl; i0 = e; }
            else if (nl > t1) { t2 = t1; t1 = nl; i1 = e; }
            else if (nl > t2) { t2 = nl; }
        }
        // softmax over top-2
        float e1  = expf(t1 - t0);
        float inv = 1.0f / (1.0f + e1);
        float g0 = inv, g1 = e1 * inv;

        outIdx [(size_t)grow * 2 + 0] = (float)i0;
        outIdx [(size_t)grow * 2 + 1] = (float)i1;
        outGate[(size_t)grow * 2 + 0] = g0;
        outGate[(size_t)grow * 2 + 1] = g1;

        atomicAdd(&impPart[i0], g0);
        atomicAdd(&impPart[i1], g1);

        // load: thr_in = 3rd largest (t2), thr_out = 2nd largest (t1)
        for (int ee = 0; ee < EEXP; ++ee) {
            int   e  = (ee + r) & (EEXP - 1);   // rotate: conflict-free LDS atomics
            float c  = ldsClean[w][r][e];
            float s  = ldsStd[w][r][e];
            float nl = c + nrow[e] * s;
            float thr = (nl > t2) ? t2 : t1;
            float z   = (c - thr) / s;
            float p   = 0.5f * erfcf(-z * 0.70710678118654752f);  // normcdf(z)
            atomicAdd(&loadPart[e], p);
        }
    }
    __syncthreads();

    if (tid < EEXP) {
        atomicAdd(&gImp[tid],  impPart[tid]);
        atomicAdd(&gLoad[tid], loadPart[tid]);
    }
}

// ---------------------------------------------------------------------------
// Kernel 2: loss = cv^2(importance) + cv^2(load), ddof=1.
// ---------------------------------------------------------------------------
__global__ void finalize_kernel(const float* __restrict__ gImp,
                                const float* __restrict__ gLoad,
                                float* __restrict__ lossOut) {
    if (threadIdx.x == 0 && blockIdx.x == 0) {
        float loss = 0.0f;
        for (int a = 0; a < 2; ++a) {
            const float* v = (a == 0) ? gImp : gLoad;
            float s = 0.0f;
            for (int i = 0; i < EEXP; ++i) s += v[i];
            float mu = s / (float)EEXP;
            float s2 = 0.0f;
            for (int i = 0; i < EEXP; ++i) { float d = v[i] - mu; s2 += d * d; }
            float var = s2 / (float)(EEXP - 1);
            loss += var / (mu * mu + 1e-10f);
        }
        *lossOut = loss;
    }
}

// ---------------------------------------------------------------------------
extern "C" void kernel_launch(void* const* d_in, const int* in_sizes, int n_in,
                              void* d_out, int out_size, void* d_ws, size_t ws_size,
                              hipStream_t stream) {
    const float* inp     = (const float*)d_in[0];
    const float* w_gate  = (const float*)d_in[1];
    const float* w_noise = (const float*)d_in[2];
    const float* noise   = (const float*)d_in[3];

    char*   ws    = (char*)d_ws;
    __bf16* wsG   = (__bf16*)ws;                          // 128 KB
    __bf16* wsN   = (__bf16*)(ws + (size_t)EEXP * DDIM * 2);
    float*  gImp  = (float*)(ws + 2 * (size_t)EEXP * DDIM * 2);
    float*  gLoad = gImp + EEXP;

    float* outF    = (float*)d_out;
    float* outIdx  = outF;                                // N*2 indices (as float)
    float* outGate = outF + (size_t)NROWS * 2;            // N*2 gates
    float* lossOut = outF + (size_t)NROWS * 4;            // scalar loss

    prep_kernel<<<(DDIM * EEXP + 255) / 256, 256, 0, stream>>>(
        w_gate, w_noise, wsG, wsN, gImp);

    moe_gate_kernel<<<NROWS / 128, 256, 0, stream>>>(
        inp, noise, wsG, wsN, outIdx, outGate, gImp, gLoad);

    finalize_kernel<<<1, 64, 0, stream>>>(gImp, gLoad, lossOut);
}
